// UnitGCN_59098749992967
// MI455X (gfx1250) — matrix operand order repaired and checked by reading, not compiled, not run on hardware
//
#include <hip/hip_runtime.h>

// ---------------------------------------------------------------------------
// AGCN unit block for MI455X (gfx1250, wave32, WMMA + TDM).
// All contractions run on v_wmma_f32_16x16x32_bf16 (f32 accumulate).
// k_conv_ab stages its x-tile with tensor_load_to_lds (Tensor Data Mover).
// Workspace: ab (bf16, 92.16 MB) + att (bf16, 384 KB).
// ---------------------------------------------------------------------------

#define N_  64
#define C_  64
#define T_  300
#define V_  25
#define S_  3
#define IC_ 16
#define OC_ 64
#define TV_ 7500            // T*V
#define ROWS_AB 96          // 2*S*IC (a rows 0..47, b rows 48..95)
#define KDIM 4800           // IC*T
#define TB  4               // t-tile in fused kernel (300 = 75*4 exact)

typedef __bf16 bf16;
typedef bf16  v16bf __attribute__((ext_vector_type(16)));
typedef float v8f   __attribute__((ext_vector_type(8)));
typedef unsigned int u32;
typedef u32 u32x4 __attribute__((ext_vector_type(4)));
typedef u32 u32x8 __attribute__((ext_vector_type(8)));

static __device__ __forceinline__ v8f zero8() {
  v8f z;
#pragma unroll
  for (int i = 0; i < 8; ++i) z[i] = 0.0f;
  return z;
}

// 16-bit A-matrix (16x32) fragment: K-base for VGPR-pair j, lane-half h.
// VGPR j: K = 16h + 2j (contiguous, matches B fragments)
static __device__ __forceinline__ int a_kbase(int j, int h) {
  return 16 * h + 2 * j;
}

// ===========================================================================
// Kernel 1: ab[n] (96x7500) = [wa;wb] (96x64) @ X_n (64x7500), + bias -> bf16
// x-tile (64 rows x 128 f32 cols) is DMA'd into LDS by the Tensor Data Mover,
// then converted to a K-contiguous bf16 tile so B fragments are ds_load_b128.
// grid (ceil(7500/128), N), block 256 (8 waves; wave = one 16-col stripe)
// ===========================================================================
__global__ __launch_bounds__(256)
void k_conv_ab(const float* __restrict__ x,
               const float* __restrict__ wa, const float* __restrict__ ba,
               const float* __restrict__ wb, const float* __restrict__ bb,
               bf16* __restrict__ ab)
{
  __shared__ bf16  sW[ROWS_AB * 64];   // [row][k]            12 KB
  __shared__ float sBias[ROWS_AB];
  __shared__ bf16  sXT[128 * 64];      // [col][k=c]          16 KB (K-contig)
  __shared__ float sStage[64 * 128];   // TDM staging [c][col] 32 KB

  const int tid  = threadIdx.x;
  const int n    = blockIdx.y;
  const int col0 = blockIdx.x * 128;

  // ---- TDM: async 2D tile load  global(f32) -> LDS staging ----
  if (tid < 32) {                       // wave 0 only (scalar instruction)
    const float* gp = x + (size_t)n * C_ * TV_ + col0;
    unsigned long long gaddr = (unsigned long long)(uintptr_t)gp;
    u32 ldsa = (u32)(uintptr_t)(&sStage[0]);   // low 32 bits = LDS byte offset
    u32 dim0 = (u32)(TV_ - col0);              // remaining cols: OOB reads -> 0

    u32x4 g0;
    g0[0] = 1u;                                    // count=1 (valid descriptor)
    g0[1] = ldsa;                                  // lds_addr
    g0[2] = (u32)(gaddr & 0xffffffffu);            // global_addr[31:0]
    g0[3] = (u32)((gaddr >> 32) & 0x1ffffffu)      // global_addr[56:32]
          | (2u << 30);                            // type=2 ("image")

    u32x8 g1;
    g1[0] = (2u << 16);                            // data_size=2 -> 4 bytes
    g1[1] = (dim0 & 0xffffu) << 16;                // tensor_dim0[15:0]
    g1[2] = ((dim0 >> 16) & 0xffffu)               // tensor_dim0[31:16]
          | (64u << 16);                           // tensor_dim1[15:0] = 64 rows
    g1[3] = (128u << 16);                          // tile_dim0 = 128 cols
    g1[4] = 64u;                                   // tile_dim1 = 64 (tile_dim2=0)
    g1[5] = (u32)TV_;                              // tensor_dim0_stride[31:0]
    g1[6] = 0u;                                    // stride0[47:32] | stride1 lo
    g1[7] = 0u;

    asm volatile("tensor_load_to_lds %0, %1" :: "s"(g0), "s"(g1) : "memory");
    __builtin_amdgcn_s_wait_tensorcnt(0);
  }

  // stage stacked weights (wa flat is exactly rows 0..47, wb rows 48..95)
  for (int idx = tid; idx < ROWS_AB * 64; idx += 256) {
    int r = idx >> 6, k = idx & 63;
    float w = (r < 48) ? wa[r * 64 + k] : wb[(r - 48) * 64 + k];
    sW[idx] = (bf16)w;
  }
  if (tid < ROWS_AB) sBias[tid] = (tid < 48) ? ba[tid] : bb[tid - 48];
  __syncthreads();

  // convert staging f32 [c][col] -> bf16 K-contiguous [col][c]
  for (int idx = tid; idx < 128 * 64; idx += 256) {
    int col = idx >> 6, k = idx & 63;
    float v = (col0 + col < TV_) ? sStage[k * 128 + col] : 0.0f;
    sXT[idx] = (bf16)v;
  }
  __syncthreads();

  const int wave = tid >> 5, lane = tid & 31;
  const int lm = lane & 15, lh = lane >> 4;

  v8f acc[6];
#pragma unroll
  for (int mt = 0; mt < 6; ++mt) acc[mt] = zero8();

#pragma unroll
  for (int ks = 0; ks < 2; ++ks) {
    const int kk = ks * 32;
    v16bf bfrag;                     // B (32x16): lanes0-15 K 0..15 (+kk)
    const int cb = (wave * 16 + lm) * 64;
#pragma unroll
    for (int j = 0; j < 8; ++j) {
      int k = kk + lh * 16 + 2 * j;
      bfrag[2 * j]     = sXT[cb + k];
      bfrag[2 * j + 1] = sXT[cb + k + 1];
    }
#pragma unroll
    for (int mt = 0; mt < 6; ++mt) {
      v16bf afrag;
      int m = mt * 16 + lm;
#pragma unroll
      for (int j = 0; j < 8; ++j) {
        int kb = kk + a_kbase(j, lh);
        afrag[2 * j]     = sW[m * 64 + kb];
        afrag[2 * j + 1] = sW[m * 64 + kb + 1];
      }
      acc[mt] = __builtin_amdgcn_wmma_f32_16x16x32_bf16(
          false, afrag, false, bfrag, (short)0, acc[mt], false, false);
    }
  }

  bf16* abn = ab + (size_t)n * ROWS_AB * TV_;
#pragma unroll
  for (int mt = 0; mt < 6; ++mt) {
#pragma unroll
    for (int r = 0; r < 8; ++r) {
      int row = mt * 16 + r + 8 * lh;      // D: VGPR r <-> M = r + 8*half
      int col = col0 + wave * 16 + lm;
      if (col < TV_) abn[row * TV_ + col] = (bf16)(acc[mt][r] + sBias[row]);
    }
  }
}

// ===========================================================================
// Kernel 2: per (s,n): scores(25x25) = a^T b / 4800 ; softmax over v ; +A+PA
// Output att TRANSPOSED and padded to 32x32 bf16: attw[n][s][u][v].
// grid (S, N), block 128 (4 waves split K=4800)
// ===========================================================================
__global__ __launch_bounds__(128)
void k_att(const bf16* __restrict__ ab,
           const float* __restrict__ A, const float* __restrict__ PA,
           bf16* __restrict__ attw)
{
  __shared__ float sRed[4 * 32 * 32];
  __shared__ float sSc[32 * 32];

  const int tid = threadIdx.x;
  const int s = blockIdx.x, n = blockIdx.y;
  const int wave = tid >> 5, lane = tid & 31;
  const int lm = lane & 15, lh = lane >> 4;

  const bf16* abn = ab + (size_t)n * ROWS_AB * TV_;
  const int arow0 = s * IC_;
  const int brow0 = 48 + s * IC_;

  v8f acc[2][2];
#pragma unroll
  for (int vt = 0; vt < 2; ++vt)
#pragma unroll
    for (int ut = 0; ut < 2; ++ut) acc[vt][ut] = zero8();

  for (int kk = wave * 32; kk < KDIM; kk += 128) {
    v16bf af[2], bg[2];
#pragma unroll
    for (int vt = 0; vt < 2; ++vt) {
      int v = vt * 16 + lm; if (v >= V_) v = 0;   // clamp: garbage rows unused
#pragma unroll
      for (int j = 0; j < 8; ++j) {
        int kb = kk + a_kbase(j, lh);
        int ic0 = kb / T_,        t0 = kb - ic0 * T_;
        int ic1 = (kb + 1) / T_,  t1 = (kb + 1) - ic1 * T_;
        af[vt][2 * j]     = abn[(arow0 + ic0) * TV_ + t0 * V_ + v];
        af[vt][2 * j + 1] = abn[(arow0 + ic1) * TV_ + t1 * V_ + v];
      }
    }
#pragma unroll
    for (int ut = 0; ut < 2; ++ut) {
      int u = ut * 16 + lm; if (u >= V_) u = 0;
#pragma unroll
      for (int j = 0; j < 8; ++j) {
        int k0 = kk + lh * 16 + 2 * j;
        int ic0 = k0 / T_,        t0 = k0 - ic0 * T_;
        int ic1 = (k0 + 1) / T_,  t1 = (k0 + 1) - ic1 * T_;
        bg[ut][2 * j]     = abn[(brow0 + ic0) * TV_ + t0 * V_ + u];
        bg[ut][2 * j + 1] = abn[(brow0 + ic1) * TV_ + t1 * V_ + u];
      }
    }
#pragma unroll
    for (int vt = 0; vt < 2; ++vt)
#pragma unroll
      for (int ut = 0; ut < 2; ++ut)
        acc[vt][ut] = __builtin_amdgcn_wmma_f32_16x16x32_bf16(
            false, af[vt], false, bg[ut], (short)0, acc[vt][ut], false, false);
  }

#pragma unroll
  for (int vt = 0; vt < 2; ++vt)
#pragma unroll
    for (int ut = 0; ut < 2; ++ut)
#pragma unroll
      for (int r = 0; r < 8; ++r) {
        int v = vt * 16 + r + 8 * lh;
        int u = ut * 16 + lm;
        sRed[wave * 1024 + v * 32 + u] = acc[vt][ut][r];
      }
  __syncthreads();

  for (int idx = tid; idx < 1024; idx += 128)
    sSc[idx] = sRed[idx] + sRed[1024 + idx] + sRed[2048 + idx] + sRed[3072 + idx];
  __syncthreads();

  if (tid < V_) {                       // one thread per column u: softmax over v
    const float inv = 1.0f / (float)KDIM;
    float mx = -3.4e38f;
    for (int v = 0; v < V_; ++v) mx = fmaxf(mx, sSc[v * 32 + tid] * inv);
    float e[V_]; float ssum = 0.f;
    for (int v = 0; v < V_; ++v) {
      float t = __expf(sSc[v * 32 + tid] * inv - mx);
      e[v] = t; ssum += t;
    }
    float rs = 1.0f / ssum;
    for (int v = 0; v < V_; ++v)
      sSc[v * 32 + tid] = e[v] * rs + A[(s * V_ + v) * V_ + tid]
                                    + PA[(s * V_ + v) * V_ + tid];
  }
  __syncthreads();

  // store TRANSPOSED: attw[n][s][u][v] (K-contiguous for step-7 B fragments)
  bf16* an = attw + (size_t)(n * S_ + s) * 1024;
  for (int idx = tid; idx < 1024; idx += 128) {
    int u = idx >> 5, v = idx & 31;
    an[idx] = (bf16)((v < V_ && u < V_) ? sSc[v * 32 + u] : 0.0f);
  }
}

// ===========================================================================
// Kernel 3 (fused): per (n, 4-t tile):
//   for s: Y = x-tile(rows (c,tt), cols v) @ att[n,s]  -> LDS (K-contiguous)
//          Z += wd[s] @ Y(rows c, cols (tt,u))
//   out = relu( Z*bnscale + bnbias2 + x )
// grid (75, N), block 256 (8 waves).  LDS ~= 62.5 KB.
// ===========================================================================
__global__ __launch_bounds__(256)
void k_fused(const float* __restrict__ x,
             const bf16* __restrict__ attw,
             const float* __restrict__ wd, const float* __restrict__ bd,
             const float* __restrict__ gamma, const float* __restrict__ beta,
             const float* __restrict__ mean,  const float* __restrict__ var,
             float* __restrict__ out)
{
  __shared__ bf16  sXt [64 * (TB * 32)];  // [c][tt*32+v]   16 KB (A side)
  __shared__ bf16  sYT [(TB * 32) * 64];  // [tt*32+u][c]   16 KB (B side, K-contig)
  __shared__ bf16  sWd [S_ * 64 * 64];    // [s][o][c]      24 KB (A side)
  __shared__ bf16  sAttT[S_ * 32 * 32];   // [s][u][v]       6 KB (B side, K-contig)
  __shared__ float sScale[64], sBias2[64];

  const int tid = threadIdx.x;
  const int n = blockIdx.y;
  const int t0 = blockIdx.x * TB;
  const int wave = tid >> 5, lane = tid & 31;
  const int lm = lane & 15, lh = lane >> 4;

  for (int idx = tid; idx < S_ * 64 * 64; idx += 256) sWd[idx] = (bf16)wd[idx];
  {
    const bf16* an = attw + (size_t)n * S_ * 1024;
    for (int idx = tid; idx < S_ * 1024; idx += 256) sAttT[idx] = an[idx];
  }
  if (tid < 64) {
    float sc  = gamma[tid] * rsqrtf(var[tid] + 1e-5f);
    float bds = bd[tid] + bd[64 + tid] + bd[128 + tid];  // sum_s bd[s][o]
    sScale[tid] = sc;
    sBias2[tid] = (bds - mean[tid]) * sc + beta[tid];
  }
  const float* xn = x + (size_t)n * C_ * TV_;
  for (int idx = tid; idx < 64 * TB * 32; idx += 256) {
    int c = idx >> 7, rem = idx & 127;
    int tt = rem >> 5, v = rem & 31;
    float val = (v < V_) ? xn[c * TV_ + (t0 + tt) * V_ + v] : 0.0f;
    sXt[idx] = (bf16)val;
  }
  __syncthreads();

  v8f zacc[4];
#pragma unroll
  for (int i = 0; i < 4; ++i) zacc[i] = zero8();

  for (int s = 0; s < S_; ++s) {
    if (s) __syncthreads();               // sYT reuse across s
    // ---- step 7: Y = Xtile @ att[s] (K = 32 padded v) ----
#pragma unroll
    for (int i = 0; i < 4; ++i) {
      int q = wave * 4 + i;               // 32 tile-jobs: 16 Mtiles x 2 Ntiles
      int mt = q >> 1, ut = q & 1;
      v16bf afrag, bfrag;
      int mrow = mt * 16 + lm;            // m = c*TB + tt
      int c = mrow >> 2, tt = mrow & 3;
#pragma unroll
      for (int j = 0; j < 8; ++j) {
        int kb = a_kbase(j, lh);
        afrag[2 * j]     = sXt[c * 128 + tt * 32 + kb];
        afrag[2 * j + 1] = sXt[c * 128 + tt * 32 + kb + 1];
      }
      int u = ut * 16 + lm;
#pragma unroll
      for (int j = 0; j < 8; ++j) {
        int k = lh * 16 + 2 * j;
        bfrag[2 * j]     = sAttT[s * 1024 + u * 32 + k];
        bfrag[2 * j + 1] = sAttT[s * 1024 + u * 32 + k + 1];
      }
      v8f y = __builtin_amdgcn_wmma_f32_16x16x32_bf16(
          false, afrag, false, bfrag, (short)0, zero8(), false, false);
#pragma unroll
      for (int r = 0; r < 8; ++r) {
        int mr = mt * 16 + r + 8 * lh;
        int cc = mr >> 2, t2 = mr & 3;
        sYT[(t2 * 32 + u) * 64 + cc] = (bf16)y[r];
      }
    }
    __syncthreads();
    // ---- step 8: Z += wd[s] @ Y (K = 64) ----
#pragma unroll
    for (int i = 0; i < 4; ++i) {
      int q = wave * 4 + i;               // 32 jobs: 4 Mtiles(o) x 8 Ntiles(col)
      int mo = q >> 3, nc = q & 7;
#pragma unroll
      for (int ks = 0; ks < 2; ++ks) {
        int kk = ks * 32;
        v16bf afrag, bfrag;
        int o = mo * 16 + lm;
#pragma unroll
        for (int j = 0; j < 8; ++j) {
          int kb = kk + a_kbase(j, lh);
          afrag[2 * j]     = sWd[(s * 64 + o) * 64 + kb];
          afrag[2 * j + 1] = sWd[(s * 64 + o) * 64 + kb + 1];
        }
        int col = nc * 16 + lm;
#pragma unroll
        for (int j = 0; j < 8; ++j) {
          int k = kk + lh * 16 + 2 * j;
          bfrag[2 * j]     = sYT[col * 64 + k];
          bfrag[2 * j + 1] = sYT[col * 64 + k + 1];
        }
        zacc[i] = __builtin_amdgcn_wmma_f32_16x16x32_bf16(
            false, afrag, false, bfrag, (short)0, zacc[i], false, false);
      }
    }
  }

  // ---- epilogue: BN + residual + ReLU ----
#pragma unroll
  for (int i = 0; i < 4; ++i) {
    int q = wave * 4 + i;
    int mo = q >> 3, nc = q & 7;
#pragma unroll
    for (int r = 0; r < 8; ++r) {
      int o = mo * 16 + r + 8 * lh;
      int col = nc * 16 + lm;
      int tt = col >> 5, u = col & 31;
      if (u < V_) {
        float zn = zacc[i][r] * sScale[o] + sBias2[o];
        size_t idx = ((size_t)(n * 64 + o) * T_ + (t0 + tt)) * V_ + u;
        out[idx] = fmaxf(zn + x[idx], 0.0f);
      }
    }
  }
}

// ===========================================================================
extern "C" void kernel_launch(void* const* d_in, const int* in_sizes, int n_in,
                              void* d_out, int out_size, void* d_ws, size_t ws_size,
                              hipStream_t stream) {
  (void)in_sizes; (void)n_in; (void)out_size; (void)ws_size;
  const float* x     = (const float*)d_in[0];
  const float* A     = (const float*)d_in[1];
  const float* PA    = (const float*)d_in[2];
  const float* wa    = (const float*)d_in[3];
  const float* ba    = (const float*)d_in[4];
  const float* wb    = (const float*)d_in[5];
  const float* bb    = (const float*)d_in[6];
  const float* wd    = (const float*)d_in[7];
  const float* bd    = (const float*)d_in[8];
  const float* gamma = (const float*)d_in[9];
  const float* beta  = (const float*)d_in[10];
  const float* mean  = (const float*)d_in[11];
  const float* var   = (const float*)d_in[12];
  float* out = (float*)d_out;

  const size_t AB_BYTES = (size_t)N_ * ROWS_AB * TV_ * sizeof(bf16); // 92.16 MB
  bf16* ab   = (bf16*)d_ws;
  bf16* attw = (bf16*)((char*)d_ws + AB_BYTES);                      // 384 KB

  k_conv_ab<<<dim3((TV_ + 127) / 128, N_), 256, 0, stream>>>(x, wa, ba, wb, bb, ab);
  k_att    <<<dim3(S_, N_),               128, 0, stream>>>(ab, A, PA, attw);
  k_fused  <<<dim3(T_ / TB, N_),          256, 0, stream>>>(x, attw, wd, bd,
                                                            gamma, beta, mean, var, out);
}
